// MultiHeadAttention_44813688766612
// MI455X (gfx1250) — matrix operand order repaired and checked by reading, not compiled, not run on hardware
//
#include <hip/hip_runtime.h>
#include <hip/hip_bf16.h>
#include <math.h>

// Problem constants (from reference)
#define Bn 4
#define Sn 2048
#define Dn 1024
#define Hn 16
#define DEPTHn 64

typedef __attribute__((ext_vector_type(16))) __bf16 v16bf;
typedef __attribute__((ext_vector_type(2)))  __bf16 v2bf;
typedef __attribute__((ext_vector_type(8)))  float  v8f;
typedef __attribute__((ext_vector_type(4)))  int    v4i;

union Frag {
    v16bf v;
    uint4 q[2];
    unsigned int u32[8];
    unsigned short u16[16];
};

__device__ __forceinline__ unsigned short f2bf(float f) {
    unsigned u = __float_as_uint(f);
    u += 0x7FFFu + ((u >> 16) & 1u);   // round-to-nearest-even
    return (unsigned short)(u >> 16);
}

#if __has_builtin(__builtin_amdgcn_cvt_pk_bf16_f32)
__device__ __forceinline__ unsigned pack2(float lo, float hi) {
    union { v2bf v; unsigned u; } c;
    c.v = __builtin_amdgcn_cvt_pk_bf16_f32(lo, hi);
    return c.u;
}
#else
__device__ __forceinline__ unsigned pack2(float lo, float hi) {
    return (unsigned)f2bf(lo) | ((unsigned)f2bf(hi) << 16);
}
#endif

// Address-space helpers for async global->LDS copies.
#if __has_builtin(__builtin_amdgcn_global_load_async_to_lds_b128)
#define USE_ASYNC_LDS 1
typedef __attribute__((address_space(1))) v4i gv4i;   // global int4
typedef __attribute__((address_space(3))) v4i lv4i;   // LDS int4

__device__ __forceinline__ gv4i* to_global(const void* p) {
    return (gv4i*)(unsigned long long)(uintptr_t)p;
}
__device__ __forceinline__ lv4i* to_lds(void* p) {
    // Generic LDS address: low 32 bits are the LDS byte offset (ISA 10.2).
    return (lv4i*)(unsigned)(uintptr_t)p;
}
__device__ __forceinline__ void wait_async0() {
#if __has_builtin(__builtin_amdgcn_s_wait_asynccnt)
    __builtin_amdgcn_s_wait_asynccnt(0);
#else
    asm volatile("s_wait_asynccnt 0" ::: "memory");
#endif
}
#else
#define USE_ASYNC_LDS 0
#endif

// ---------------------------------------------------------------------------
// K0: W (fp32, [K][N] row-major) -> Wt (bf16, [N][K] row-major)
// ---------------------------------------------------------------------------
__global__ void transpose_w_bf16(const float* __restrict__ W,
                                 unsigned short* __restrict__ Wt) {
    int idx = blockIdx.x * blockDim.x + threadIdx.x;   // 1M elements
    int n = idx >> 10;
    int k = idx & 1023;
    Wt[(size_t)n * Dn + k] = f2bf(W[(size_t)k * Dn + n]);
}

// ---------------------------------------------------------------------------
// K1/K5: C[M x 1024] = A[M x 1024] @ W + bias
//   A: fp32 (a_is_f32=1) or bf16 (a_is_f32=0), row-major, lda=1024
//   Wt: bf16 [N][K] (pre-transposed); tile staged to LDS via async copies
//   out_mode 0: fp32 row-major [M][1024]
//   out_mode 1: bf16 head-split   [b][h][s][64]
//   out_mode 2: bf16 head-split-T [b][h][64][s]
// block = 128 threads (4 waves); block tile 64(M) x 64(N); wave tile 16x64
// ---------------------------------------------------------------------------
#define WSTRIDE 40   // LDS row pitch in shorts (80B): conflict-free b128 reads

__global__ __launch_bounds__(128)
void gemm_wmma(const void* __restrict__ Ap, int a_is_f32,
               const unsigned short* __restrict__ Wt,
               const float* __restrict__ bias,
               void* __restrict__ Out, int out_mode) {
    __shared__ unsigned short wtile[64 * WSTRIDE];

    const int lane = threadIdx.x & 31;
    const int wave = threadIdx.x >> 5;
    const int half = lane >> 4;
    const int ln   = lane & 15;
    const int Mbase = blockIdx.y * 64 + wave * 16;
    const int Nbase = blockIdx.x * 64;
    const int m = Mbase + ln;

    // Per-thread staging coordinates: 128 threads x 32B = 4KB tile (64x32 bf16)
    const int sn = threadIdx.x >> 1;          // 0..63 : N row within tile
    const int sk = (threadIdx.x & 1) * 16;    // 0/16  : K offset within tile

    v8f acc[4] = {};
    for (int kb = 0; kb < Dn; kb += 32) {
        __syncthreads();   // previous tile fully consumed
        {
            const unsigned short* src = Wt + (size_t)(Nbase + sn) * Dn + kb + sk;
            unsigned short* dst = &wtile[sn * WSTRIDE + sk];
#if USE_ASYNC_LDS
            __builtin_amdgcn_global_load_async_to_lds_b128(to_global(src), to_lds(dst), 0, 0);
            __builtin_amdgcn_global_load_async_to_lds_b128(to_global(src + 8), to_lds(dst + 8), 0, 0);
            wait_async0();
#else
            *(uint4*)dst       = *(const uint4*)src;
            *(uint4*)(dst + 8) = *(const uint4*)(src + 8);
#endif
        }
        __syncthreads();

        Frag a;
        if (a_is_f32) {
            const float* A = (const float*)Ap + (size_t)m * Dn + kb + 8 * half;
            float4 f0 = *(const float4*)(A);
            float4 f1 = *(const float4*)(A + 4);
            float4 f2 = *(const float4*)(A + 16);
            float4 f3 = *(const float4*)(A + 20);
            a.u32[0] = pack2(f0.x, f0.y); a.u32[1] = pack2(f0.z, f0.w);
            a.u32[2] = pack2(f1.x, f1.y); a.u32[3] = pack2(f1.z, f1.w);
            a.u32[4] = pack2(f2.x, f2.y); a.u32[5] = pack2(f2.z, f2.w);
            a.u32[6] = pack2(f3.x, f3.y); a.u32[7] = pack2(f3.z, f3.w);
        } else {
            const unsigned short* A = (const unsigned short*)Ap + (size_t)m * Dn + kb + 8 * half;
            a.q[0] = *(const uint4*)(A);
            a.q[1] = *(const uint4*)(A + 16);
        }
        for (int nt = 0; nt < 4; ++nt) {
            const unsigned short* w = &wtile[(nt * 16 + ln) * WSTRIDE + 8 * half];
            Frag b;
            b.q[0] = *(const uint4*)(w);
            b.q[1] = *(const uint4*)(w + 16);
            acc[nt] = __builtin_amdgcn_wmma_f32_16x16x32_bf16(
                false, a.v, false, b.v, (short)0, acc[nt], false, false);
        }
    }

    for (int nt = 0; nt < 4; ++nt) {
        int col = Nbase + nt * 16 + ln;
        float bv = bias[col];
        for (int r = 0; r < 8; ++r) {
            int row = Mbase + r + 8 * half;
            float val = acc[nt][r] + bv;
            if (out_mode == 0) {
                ((float*)Out)[(size_t)row * Dn + col] = val;
            } else {
                int bb = row >> 11, s = row & (Sn - 1);
                int hh = col >> 6,  dd = col & 63;
                size_t bh = (size_t)(bb * Hn + hh);
                if (out_mode == 1)
                    ((unsigned short*)Out)[(bh * Sn + s) * DEPTHn + dd] = f2bf(val);
                else
                    ((unsigned short*)Out)[(bh * DEPTHn + dd) * (size_t)Sn + s] = f2bf(val);
            }
        }
    }
}

// ---------------------------------------------------------------------------
// K2: scores = (q @ k^T) * 0.125 + mask bias, written fp32 into attn region
// block = 128 threads (4 waves); block handles one (b,h, 16-row q-tile),
// waves stripe the 128 key tiles.
// ---------------------------------------------------------------------------
__global__ __launch_bounds__(128)
void scores_wmma(const unsigned short* __restrict__ qhs,
                 const unsigned short* __restrict__ khs,
                 const int* __restrict__ mask,
                 float* __restrict__ attn) {
    const int lane = threadIdx.x & 31;
    const int wave = threadIdx.x >> 5;
    const int half = lane >> 4;
    const int ln   = lane & 15;
    const int qt = blockIdx.x;     // 0..127
    const int bh = blockIdx.y;     // 0..63
    const int b  = bh >> 4;

    const unsigned short* qrow = qhs + ((size_t)bh * Sn + qt * 16 + ln) * DEPTHn;
    Frag a0, a1;
    a0.q[0] = *(const uint4*)(qrow + 8 * half);
    a0.q[1] = *(const uint4*)(qrow + 16 + 8 * half);
    a1.q[0] = *(const uint4*)(qrow + 32 + 8 * half);
    a1.q[1] = *(const uint4*)(qrow + 48 + 8 * half);

    for (int nt = wave; nt < Sn / 16; nt += 4) {
        const unsigned short* krow = khs + ((size_t)bh * Sn + nt * 16 + ln) * DEPTHn;
        Frag b0, b1;
        b0.q[0] = *(const uint4*)(krow + 8 * half);
        b0.q[1] = *(const uint4*)(krow + 16 + 8 * half);
        b1.q[0] = *(const uint4*)(krow + 32 + 8 * half);
        b1.q[1] = *(const uint4*)(krow + 48 + 8 * half);
        v8f c = {};
        c = __builtin_amdgcn_wmma_f32_16x16x32_bf16(false, a0.v, false, b0.v,
                                                    (short)0, c, false, false);
        c = __builtin_amdgcn_wmma_f32_16x16x32_bf16(false, a1.v, false, b1.v,
                                                    (short)0, c, false, false);
        int key = nt * 16 + ln;
        for (int r = 0; r < 8; ++r) {
            int qpos = qt * 16 + r + 8 * half;
            float s = c[r] * 0.125f;   // 1/sqrt(64)
            int mv = mask[((size_t)b * Sn + qpos) * Sn + key];
            if (mv == 0) s -= 1e9f;
            attn[((size_t)bh * Sn + qpos) * Sn + key] = s;
        }
    }
}

// ---------------------------------------------------------------------------
// K3: in-place row softmax over attn rows of length 2048. One block per row.
// ---------------------------------------------------------------------------
__global__ __launch_bounds__(256)
void softmax_rows(float* __restrict__ attn) {
    __shared__ float red[256];
    const int t = threadIdx.x;
    float* row = attn + (size_t)blockIdx.x * Sn;

    float4 v0 = *(float4*)(row + t * 8);
    float4 v1 = *(float4*)(row + t * 8 + 4);
    float mx = fmaxf(fmaxf(fmaxf(v0.x, v0.y), fmaxf(v0.z, v0.w)),
                     fmaxf(fmaxf(v1.x, v1.y), fmaxf(v1.z, v1.w)));
    red[t] = mx;
    __syncthreads();
    for (int s = 128; s > 0; s >>= 1) {
        if (t < s) red[t] = fmaxf(red[t], red[t + s]);
        __syncthreads();
    }
    mx = red[0];
    __syncthreads();

    float e[8];
    e[0] = __expf(v0.x - mx); e[1] = __expf(v0.y - mx);
    e[2] = __expf(v0.z - mx); e[3] = __expf(v0.w - mx);
    e[4] = __expf(v1.x - mx); e[5] = __expf(v1.y - mx);
    e[6] = __expf(v1.z - mx); e[7] = __expf(v1.w - mx);
    float sum = (e[0] + e[1] + e[2] + e[3]) + (e[4] + e[5] + e[6] + e[7]);
    red[t] = sum;
    __syncthreads();
    for (int s = 128; s > 0; s >>= 1) {
        if (t < s) red[t] += red[t + s];
        __syncthreads();
    }
    float inv = 1.0f / red[0];

    float4 o0 = make_float4(e[0] * inv, e[1] * inv, e[2] * inv, e[3] * inv);
    float4 o1 = make_float4(e[4] * inv, e[5] * inv, e[6] * inv, e[7] * inv);
    *(float4*)(row + t * 8) = o0;
    *(float4*)(row + t * 8 + 4) = o1;
}

// ---------------------------------------------------------------------------
// K4: ctx[b][s][h*64+dd] = attn @ v  (A = fp32 attn -> bf16 frags, B = vT bf16)
// block = 1 wave; block handles one (b,h, 16-row q-tile), full 64-wide depth.
// ---------------------------------------------------------------------------
__global__ __launch_bounds__(32)
void ctx_wmma(const float* __restrict__ attn,
              const unsigned short* __restrict__ vT,
              unsigned short* __restrict__ ctx) {
    const int lane = threadIdx.x & 31;
    const int half = lane >> 4;
    const int ln   = lane & 15;
    const int qt = blockIdx.x;   // 0..127
    const int bh = blockIdx.y;   // 0..63
    const int b  = bh >> 4, h = bh & 15;

    v8f acc[4] = {};
    const float* arow = attn + ((size_t)bh * Sn + qt * 16 + ln) * Sn;
    for (int kc = 0; kc < Sn; kc += 32) {
        Frag a;
        const float* p = arow + kc + 8 * half;
        float4 f0 = *(const float4*)(p);
        float4 f1 = *(const float4*)(p + 4);
        float4 f2 = *(const float4*)(p + 16);
        float4 f3 = *(const float4*)(p + 20);
        a.u32[0] = pack2(f0.x, f0.y); a.u32[1] = pack2(f0.z, f0.w);
        a.u32[2] = pack2(f1.x, f1.y); a.u32[3] = pack2(f1.z, f1.w);
        a.u32[4] = pack2(f2.x, f2.y); a.u32[5] = pack2(f2.z, f2.w);
        a.u32[6] = pack2(f3.x, f3.y); a.u32[7] = pack2(f3.z, f3.w);
        for (int dt = 0; dt < 4; ++dt) {
            const unsigned short* vrow =
                vT + ((size_t)bh * DEPTHn + dt * 16 + ln) * Sn + kc + 8 * half;
            Frag bb;
            bb.q[0] = *(const uint4*)(vrow);
            bb.q[1] = *(const uint4*)(vrow + 16);
            acc[dt] = __builtin_amdgcn_wmma_f32_16x16x32_bf16(
                false, a.v, false, bb.v, (short)0, acc[dt], false, false);
        }
    }
    for (int dt = 0; dt < 4; ++dt) {
        for (int r = 0; r < 8; ++r) {
            int qpos = qt * 16 + r + 8 * half;
            ctx[((size_t)b * Sn + qpos) * Dn + h * DEPTHn + dt * 16 + ln] =
                f2bf(acc[dt][r]);
        }
    }
}

// ---------------------------------------------------------------------------
extern "C" void kernel_launch(void* const* d_in, const int* in_sizes, int n_in,
                              void* d_out, int out_size, void* d_ws, size_t ws_size,
                              hipStream_t stream) {
    const float* Q  = (const float*)d_in[0];
    const float* K  = (const float*)d_in[1];
    const float* V  = (const float*)d_in[2];
    const int*   mk = (const int*)d_in[3];
    const float* Wq = (const float*)d_in[4];  const float* bq = (const float*)d_in[5];
    const float* Wk = (const float*)d_in[6];  const float* bk = (const float*)d_in[7];
    const float* Wv = (const float*)d_in[8];  const float* bv = (const float*)d_in[9];
    const float* Wo = (const float*)d_in[10]; const float* bo = (const float*)d_in[11];

    float* out  = (float*)d_out;
    float* attn = out + (size_t)Bn * Sn * Dn;   // attn follows out in d_out

    // Workspace carve-up (bf16): 4 x WT (2MB) + q + k + vT + ctx (16MB each) = 72MB
    unsigned short* WqT = (unsigned short*)d_ws;
    unsigned short* WkT = WqT + (size_t)Dn * Dn;
    unsigned short* WvT = WkT + (size_t)Dn * Dn;
    unsigned short* WoT = WvT + (size_t)Dn * Dn;
    unsigned short* qhs = WoT + (size_t)Dn * Dn;
    unsigned short* khs = qhs + (size_t)Bn * Sn * Dn;
    unsigned short* vT  = khs + (size_t)Bn * Sn * Dn;
    unsigned short* ctx = vT  + (size_t)Bn * Sn * Dn;

    // K0: weight transpose + bf16 convert
    transpose_w_bf16<<<4096, 256, 0, stream>>>(Wq, WqT);
    transpose_w_bf16<<<4096, 256, 0, stream>>>(Wk, WkT);
    transpose_w_bf16<<<4096, 256, 0, stream>>>(Wv, WvT);
    transpose_w_bf16<<<4096, 256, 0, stream>>>(Wo, WoT);

    // K1: projections (M = B*S = 8192)
    dim3 ggrid(Dn / 64, (Bn * Sn) / 64);
    gemm_wmma<<<ggrid, 128, 0, stream>>>(Q, 1, WqT, bq, qhs, 1);
    gemm_wmma<<<ggrid, 128, 0, stream>>>(K, 1, WkT, bk, khs, 1);
    gemm_wmma<<<ggrid, 128, 0, stream>>>(V, 1, WvT, bv, vT, 2);

    // K2: masked scaled scores -> attn (fp32)
    scores_wmma<<<dim3(Sn / 16, Bn * Hn), 128, 0, stream>>>(qhs, khs, mk, attn);

    // K3: row softmax in place
    softmax_rows<<<Bn * Hn * Sn, 256, 0, stream>>>(attn);

    // K4: ctx = attn @ v
    ctx_wmma<<<dim3(Sn / 16, Bn * Hn), 32, 0, stream>>>(attn, vT, ctx);

    // K5: out = ctx @ Wo + bo (fp32 out)
    gemm_wmma<<<ggrid, 128, 0, stream>>>(ctx, 0, WoT, bo, out, 0);
}
